// SpatialConvSSM_13091060318333
// MI455X (gfx1250) — matrix operand order repaired and checked by reading, not compiled, not run on hardware
//
#include <hip/hip_runtime.h>

// ---------------------------------------------------------------------------
// SpatialConvSSM on MI455X (gfx1250):
//   h_T = sum_{t=0..7} A*^{conv t} (x conv B),  A* = 0.9*tanh(A_kernel)
// 8 depthwise-conv passes (Bx pass + 7 recurrence passes), each mapped onto
// v_wmma_f32_16x16x32_f16 via banded weight matrices (7 WMMAs per 16x16 tile
// per channel). Intermediates ping-pong through L2-resident f32 buffers
// (tensor = 134MB, global L2 = 192MB); WMMA A-operands staged f32->f16 in LDS,
// D tiles routed back through LDS for fully coalesced float4 global I/O.
// ---------------------------------------------------------------------------

typedef __attribute__((ext_vector_type(16))) _Float16 v16h;
typedef __attribute__((ext_vector_type(8)))  float    v8f;

#define BATCH 8
#define HH    128
#define WW    128
#define CC    256
#define KS    7

// LDS staging: 16 channels x 22 rows x 32 cols (f16), row pitch 40 f16 (80B)
// -> b128 fragment reads are 16B aligned and bank-conflict free.
#define LROWS  22
#define LCOLS  32
#define LPITCH 40
#define LCH    (LROWS * LPITCH)   // 880 f16 per channel

// Output staging (f32 view of the same LDS): 16 ch x 16 rows, row pitch 20,
// channel stride 340 f32 -> conflict-free writes {m,32+m} and reads
// {q*16 + i*20 + col}.  16*340*4 = 21760B <= 28160B buffer.
#define SOPITCH 20
#define SOCH    340

// Fragment storage: per (channel, dy): 32 lanes x 16 f16 = 512 f16 (1KB),
// laid out exactly as the B-operand VGPR file (lane-major, v16h per lane).
#define FRAG_PER_CDY 512

// ---------------------------------------------------------------------------
// Precompute banded B-matrices for both kernels.
//   B_dy[k][n] = wt[dy][k-n-5] if 0 <= k-n-5 <= 6 else 0   (k in [0,32))
// B-operand k order used here (contiguous, same as A): lane = n + 16*hh holds K = hh*16+e.
// ---------------------------------------------------------------------------
__global__ __launch_bounds__(224) void make_frags(const float* __restrict__ Ak,
                                                  const float* __restrict__ Bk,
                                                  _Float16* __restrict__ fragA,
                                                  _Float16* __restrict__ fragB) {
  int ch  = blockIdx.x;            // 0..255
  int tid = threadIdx.x;           // 0..223
  int dy   = tid >> 5;             // 0..6
  int lane = tid & 31;
  int n  = lane & 15;
  int hh = lane >> 4;

  float wA[KS], wB[KS];
#pragma unroll
  for (int dx = 0; dx < KS; ++dx) {
    float a = Ak[(dy * KS + dx) * CC + ch];
    wA[dx] = 0.9f * tanhf(a);
    wB[dx] = Bk[(dy * KS + dx) * CC + ch];
  }

  size_t base = ((size_t)ch * KS + dy) * FRAG_PER_CDY + (size_t)lane * 16;
  _Float16* oA = fragA + base;
  _Float16* oB = fragB + base;
#pragma unroll
  for (int e = 0; e < 16; ++e) {
    int k = hh * 16 + e;
    int d = k - n - 5;
    bool in_band = (d >= 0) && (d < KS);
    int ds = in_band ? d : 0;
    oA[e] = in_band ? (_Float16)wA[ds] : (_Float16)0.0f;
    oB[e] = in_band ? (_Float16)wB[ds] : (_Float16)0.0f;
  }
}

// ---------------------------------------------------------------------------
// One depthwise conv pass: out = dwconv(in, frag-weights) + drive(optional).
// Block = 256 threads = 8 waves; handles (batch b, 16 channels, 16x16 tile).
// Wave w handles channels {2w, 2w+1} with two interleaved WMMA accumulators.
// ---------------------------------------------------------------------------
__global__ __launch_bounds__(256) void conv_step(const float* __restrict__ in,
                                                 const float* __restrict__ drive,
                                                 float* __restrict__ out,
                                                 const _Float16* __restrict__ frag) {
  const int b  = blockIdx.z;
  const int cb = blockIdx.y * 16;        // channel block base
  const int ty = blockIdx.x >> 3;
  const int tw = blockIdx.x & 7;
  const int y0 = ty * 16;
  const int w0 = tw * 16;

  __shared__ __align__(16) unsigned char lsraw[16 * LCH * sizeof(_Float16)]; // 28160B
  _Float16* smem = (_Float16*)lsraw;     // f16 input-staging view
  float*    sout = (float*)lsraw;        // f32 output-staging view (reused)

  const int tid  = threadIdx.x;
  const int wave = tid >> 5;
  const int lane = tid & 31;
  const int m  = lane & 15;              // A-operand row / D column index
  const int hh = lane >> 4;

  const int ch0 = wave * 2;
  const int ch1 = ch0 + 1;
  const int c0  = cb + ch0;
  const int c1  = cb + ch1;

  // ---- Hoist all banded weight fragments (loop-invariant per tile).
  // Issued before the LDS fill so their latency hides under staging.
  v16h bf0[KS], bf1[KS];
#pragma unroll
  for (int dy = 0; dy < KS; ++dy) {
    bf0[dy] = *(const v16h*)(frag + ((size_t)c0 * KS + dy) * FRAG_PER_CDY + (size_t)lane * 16);
    bf1[dy] = *(const v16h*)(frag + ((size_t)c1 * KS + dy) * FRAG_PER_CDY + (size_t)lane * 16);
  }

  // ---- Stage halo window to LDS as f16.
  // Thread loads a float4 (4 consecutive channels); lanes 0..3 cover the 16
  // channels of one (y,w) position -> 64B contiguous global bursts.
  {
    const int q = tid & 3;               // channel quad
    for (int j = tid >> 2; j < LROWS * LCOLS; j += 64) {   // 11 iterations
      int row = j >> 5;                  // /32
      int col = j & 31;
      int y = y0 - 3 + row;
      int w = w0 - 8 + col;
      float4 v = make_float4(0.0f, 0.0f, 0.0f, 0.0f);
      if ((unsigned)y < HH && (unsigned)w < WW)
        v = *(const float4*)&in[(((size_t)b * HH + y) * WW + w) * CC + cb + q * 4];
      _Float16* d = &smem[(q * 4) * LCH + row * LPITCH + col];
      d[0 * LCH] = (_Float16)v.x;
      d[1 * LCH] = (_Float16)v.y;
      d[2 * LCH] = (_Float16)v.z;
      d[3 * LCH] = (_Float16)v.w;
    }
  }
  __syncthreads();

  v8f acc0 = {};
  v8f acc1 = {};

  union AFrag { uint4 q[2]; v16h v; };

#pragma unroll
  for (int dy = 0; dy < KS; ++dy) {
    // A operand: 16x32 f16 window, contiguous lane layout (K=hh*16+e)
    const _Float16* base0 = &smem[ch0 * LCH + (m + dy) * LPITCH + hh * 16];
    const _Float16* base1 = &smem[ch1 * LCH + (m + dy) * LPITCH + hh * 16];
    AFrag a0, a1;
    a0.q[0] = *(const uint4*)(base0);
    a0.q[1] = *(const uint4*)(base0 + 8);
    a1.q[0] = *(const uint4*)(base1);
    a1.q[1] = *(const uint4*)(base1 + 8);

    acc0 = __builtin_amdgcn_wmma_f32_16x16x32_f16(false, a0.v, false, bf0[dy],
                                                  (short)0, acc0, false, false);
    acc1 = __builtin_amdgcn_wmma_f32_16x16x32_f16(false, a1.v, false, bf1[dy],
                                                  (short)0, acc1, false, false);
  }

  // ---- D tile -> LDS (f32 view), then coalesced drive-add + float4 store ----
  __syncthreads();                       // all A-fragment reads complete
#pragma unroll
  for (int r = 0; r < 8; ++r) {
    sout[ch0 * SOCH + (r + 8 * hh) * SOPITCH + m] = acc0[r];
    sout[ch1 * SOCH + (r + 8 * hh) * SOPITCH + m] = acc1[r];
  }
  __syncthreads();

  {
    const size_t tbase = (((size_t)b * HH + y0) * WW + w0) * CC + cb;
    const int q = tid & 3;
    float* op = out + tbase + q * 4;
    if (drive) {
      const float* dp = drive + tbase + q * 4;
      for (int j = tid >> 2; j < 256; j += 64) {           // 4 iterations
        int row = j >> 4;
        int col = j & 15;
        int soff = row * SOPITCH + col;
        int goff = (row * WW + col) * CC;
        float4 d4 = *(const float4*)(dp + goff);
        float4 v;
        v.x = sout[(q * 4 + 0) * SOCH + soff] + d4.x;
        v.y = sout[(q * 4 + 1) * SOCH + soff] + d4.y;
        v.z = sout[(q * 4 + 2) * SOCH + soff] + d4.z;
        v.w = sout[(q * 4 + 3) * SOCH + soff] + d4.w;
        *(float4*)(op + goff) = v;
      }
    } else {
      for (int j = tid >> 2; j < 256; j += 64) {
        int row = j >> 4;
        int col = j & 15;
        int soff = row * SOPITCH + col;
        int goff = (row * WW + col) * CC;
        float4 v;
        v.x = sout[(q * 4 + 0) * SOCH + soff];
        v.y = sout[(q * 4 + 1) * SOCH + soff];
        v.z = sout[(q * 4 + 2) * SOCH + soff];
        v.w = sout[(q * 4 + 3) * SOCH + soff];
        *(float4*)(op + goff) = v;
      }
    }
  }
}

// ---------------------------------------------------------------------------
// Launch: make fragments, compute Bx, then 7 recurrence passes.
// d_ws: [Bx 134MB][ping 134MB][fragA 1.75MB][fragB 1.75MB]; d_out doubles as
// the pong buffer with parity so that h8 lands in d_out.
// ---------------------------------------------------------------------------
extern "C" void kernel_launch(void* const* d_in, const int* in_sizes, int n_in,
                              void* d_out, int out_size, void* d_ws, size_t ws_size,
                              hipStream_t stream) {
  const float* x  = (const float*)d_in[0];
  const float* Ak = (const float*)d_in[1];
  const float* Bk = (const float*)d_in[2];
  float* outp = (float*)d_out;

  const size_t NE = (size_t)BATCH * HH * WW * CC;          // 33,554,432
  float* W0 = (float*)d_ws;                                // Bx drive
  float* W1 = W0 + NE;                                     // ping
  _Float16* fragA = (_Float16*)((char*)d_ws + 2 * NE * sizeof(float));
  _Float16* fragB = fragA + (size_t)CC * KS * FRAG_PER_CDY;

  make_frags<<<dim3(CC), dim3(224), 0, stream>>>(Ak, Bk, fragA, fragB);

  dim3 grid((HH / 16) * (WW / 16), CC / 16, BATCH);        // (64,16,8)
  dim3 blk(256);

  // h1 = Bx
  conv_step<<<grid, blk, 0, stream>>>(x, nullptr, W0, fragB);
  // h2..h8 = A*h + Bx  (final pass writes d_out)
  conv_step<<<grid, blk, 0, stream>>>(W0,   W0, outp, fragA);  // h2
  conv_step<<<grid, blk, 0, stream>>>(outp, W0, W1,   fragA);  // h3
  conv_step<<<grid, blk, 0, stream>>>(W1,   W0, outp, fragA);  // h4
  conv_step<<<grid, blk, 0, stream>>>(outp, W0, W1,   fragA);  // h5
  conv_step<<<grid, blk, 0, stream>>>(W1,   W0, outp, fragA);  // h6
  conv_step<<<grid, blk, 0, stream>>>(outp, W0, W1,   fragA);  // h7
  conv_step<<<grid, blk, 0, stream>>>(W1,   W0, outp, fragA);  // h8 -> d_out
}
